// SelfAttentionCouplingBlock_54339926229634
// MI455X (gfx1250) — compile-verified, matrix operand order checked by reading, not run on hardware
//
#include <hip/hip_runtime.h>

#define BB 8
#define TT 1024
#define CC 512
#define HH 8
#define EE 256            // head dim == E
#define MROWS (BB * TT)   // 8192
#define NO (HH * EE)      // 2048

typedef __attribute__((ext_vector_type(16))) _Float16 v16h;
typedef __attribute__((ext_vector_type(8)))  float    v8f;
typedef _Float16 h2 __attribute__((ext_vector_type(2)));
typedef _Float16 h8 __attribute__((ext_vector_type(8)));

// K-offset of the even element of VGPR-pair j for an A/B 16x32 f16 fragment.
// Lanes 0-15 (hh=0): VGPR0..3 -> K 0..7, VGPR4..7 -> K 16..23
// Lanes 16-31 (hh=1): VGPR0..3 -> K 8..15, VGPR4..7 -> K 24..31
__device__ __forceinline__ int ak(int j, int hh) {
  return ((j >> 2) << 4) + (hh << 3) + ((j & 3) << 1);
}

__device__ __forceinline__ v8f wmma_f16(v16h a, v16h b, v8f c) {
  return __builtin_amdgcn_wmma_f32_16x16x32_f16(false, a, false, b, (short)0, c,
                                                false, false);
}

// ---------------------------------------------------------------------------
// Kernel 1: fused Q/K/V projection.  X1 (8192x256) * W^T (256x2048) -> f16.
// Each wave computes a 16x64 output strip (4 WMMA tiles) so the A fragment is
// reused 4x; grid = (512, 96), blockIdx.y selects {q,k,v} and the 64-col strip.
// q and k are pre-scaled by E^-0.25 = 0.25 so attention needs no extra scale.
// ---------------------------------------------------------------------------
__global__ __launch_bounds__(32) void qkv_proj(const float* __restrict__ x,
                                               const float* __restrict__ Wq,
                                               const float* __restrict__ Wk,
                                               const float* __restrict__ Wv,
                                               _Float16* __restrict__ Q,
                                               _Float16* __restrict__ K,
                                               _Float16* __restrict__ V) {
  const int l  = threadIdx.x;
  const int m0 = blockIdx.x << 4;
  const int nt = blockIdx.y;
  const int which = nt >> 5;          // 0=q, 1=k, 2=v
  const int n0 = (nt & 31) << 6;      // 64-wide strip
  const float* W = (which == 0) ? Wq : (which == 1) ? Wk : Wv;
  _Float16* out  = (which == 0) ? Q  : (which == 1) ? K  : V;
  const float scale = (which == 2) ? 1.0f : 0.25f;

  const int hh = l >> 4;
  const int lane15 = l & 15;
  const int mA = m0 + lane15;
  v8f acc[4];
  const v8f vzero = {};
#pragma unroll
  for (int t = 0; t < 4; ++t) acc[t] = vzero;

#pragma unroll
  for (int kb = 0; kb < EE; kb += 32) {
    v16h a;
#pragma unroll
    for (int j = 0; j < 8; ++j) {
      const int k0 = kb + ak(j, hh);
      const float* ap = &x[mA * CC + k0];       // x1 = x[..., :256]
      a[2 * j]     = (_Float16)ap[0];
      a[2 * j + 1] = (_Float16)ap[1];
    }
#pragma unroll
    for (int t = 0; t < 4; ++t) {
      v16h b;
#pragma unroll
      for (int j = 0; j < 8; ++j) {
        const int k0 = kb + ak(j, hh);
        const float* bp = &W[(n0 + (t << 4) + lane15) * EE + k0];
        b[2 * j]     = (_Float16)bp[0];
        b[2 * j + 1] = (_Float16)bp[1];
      }
      acc[t] = wmma_f16(a, b, acc[t]);
    }
  }
#pragma unroll
  for (int t = 0; t < 4; ++t) {
    const int col = n0 + (t << 4) + lane15;
#pragma unroll
    for (int r = 0; r < 8; ++r) {
      const int row = m0 + r + (hh << 3);
      out[row * NO + col] = (_Float16)(acc[t][r] * scale);
    }
  }
}

// ---------------------------------------------------------------------------
// Kernel 2: flash attention.  4 waves per workgroup; each wave owns 16 query
// rows.  K/V chunks are DOUBLE-BUFFERED in LDS and staged with the CDNA5
// async global->LDS path (global_load_async_to_lds_b128 / ASYNCcnt): while
// chunk i is consumed by 32 WMMAs per wave, chunk i+1 streams into the other
// buffer with no VGPR round-trip.  Async loads complete in order, so waiting
// ASYNCcnt<=16 after issuing the next chunk guarantees the current chunk is
// resident.
// ---------------------------------------------------------------------------
__global__ __launch_bounds__(128) void attn(const _Float16* __restrict__ Q,
                                            const _Float16* __restrict__ K,
                                            const _Float16* __restrict__ V,
                                            _Float16* __restrict__ O) {
  __shared__ __align__(16) _Float16 qtile[4][16][EE];   // 32 KB
  __shared__ __align__(16) _Float16 kvs[2][2][32][EE];  // 64 KB (2 bufs, K+V)
  __shared__ __align__(16) _Float16 pbuf[4][16][32];    //  4 KB

  const int tid = threadIdx.x;
  const int w = tid >> 5;            // wave id 0..3
  const int l = tid & 31;
  const int t64 = blockIdx.x & 15;   // 64-row query block
  const int bh = blockIdx.x >> 4;
  const int b = bh >> 3, h = bh & 7;
  const int hoff = h * EE;
  const int rbase = b * TT + (t64 << 6) + (w << 4);  // this wave's 16 rows
  const int kvrow = b * TT;

  // async-stage one 32-key K+V chunk (16 b128 instructions per wave)
  auto stage_async = [&](int nbuf, int ns0) {
#pragma unroll
    for (int i = 0; i < 8; ++i) {
      const int idx = i * 128 + tid;  // 1024 h8 per matrix
      const int r = idx >> 5, c8 = (idx & 31) << 3;
      const unsigned int lK =
          (unsigned int)(unsigned long long)&kvs[nbuf][0][r][c8];
      const unsigned int lV =
          (unsigned int)(unsigned long long)&kvs[nbuf][1][r][c8];
      const unsigned long long gK =
          (unsigned long long)&K[(kvrow + ns0 + r) * NO + hoff + c8];
      const unsigned long long gV =
          (unsigned long long)&V[(kvrow + ns0 + r) * NO + hoff + c8];
      asm volatile(
          "global_load_async_to_lds_b128 %0, %1, off\n\t"
          "global_load_async_to_lds_b128 %2, %3, off"
          :: "v"(lK), "v"(gK), "v"(lV), "v"(gV)
          : "memory");
    }
  };

  // each wave stages its own 16x256 f16 Q tile (coalesced h8 loads)
#pragma unroll
  for (int i = 0; i < 16; ++i) {
    const int idx = i * 32 + l;      // 512 h8 per tile
    const int r = idx >> 5, c8 = (idx & 31) << 3;
    *(h8*)&qtile[w][r][c8] = *(const h8*)&Q[(rbase + r) * NO + hoff + c8];
  }

  // prologue: start streaming chunk 0 into buffer 0
  stage_async(0, 0);

  const int hh = l >> 4;
  const int lane15 = l & 15;
  const v8f vzero = {};
  float mrow[8], lrow[8];
  v8f ot[16];
#pragma unroll
  for (int r = 0; r < 8; ++r) { mrow[r] = -1e30f; lrow[r] = 0.0f; }
#pragma unroll
  for (int t = 0; t < 16; ++t) ot[t] = vzero;

  for (int s0 = 0; s0 < TT; s0 += 32) {
    const int buf = (s0 >> 5) & 1;
    // kick off the next chunk into the other buffer (safe: last read of that
    // buffer finished before the trailing barrier of the previous iteration),
    // then wait until the CURRENT chunk is fully in LDS.
    if (s0 + 32 < TT) {
      stage_async(buf ^ 1, s0 + 32);
      asm volatile("s_wait_asynccnt 0x10" ::: "memory");
    } else {
      asm volatile("s_wait_asynccnt 0x0" ::: "memory");
    }
    __syncthreads();

    // ---- scores: S = Q * K^T for 32 keys (two 16x16 tiles) ----
    v8f sA = vzero, sB = vzero;
#pragma unroll
    for (int ch = 0; ch < 8; ++ch) {
      v16h qf, kf0, kf1;
#pragma unroll
      for (int j = 0; j < 8; ++j) {
        const int k0 = (ch << 5) + ak(j, hh);
        const h2 qq = *(const h2*)&qtile[w][lane15][k0];
        qf[2 * j] = qq.x; qf[2 * j + 1] = qq.y;
        const h2 kk0 = *(const h2*)&kvs[buf][0][lane15][k0];
        kf0[2 * j] = kk0.x; kf0[2 * j + 1] = kk0.y;
        const h2 kk1 = *(const h2*)&kvs[buf][0][16 + lane15][k0];
        kf1[2 * j] = kk1.x; kf1[2 * j + 1] = kk1.y;
      }
      sA = wmma_f16(qf, kf0, sA);
      sB = wmma_f16(qf, kf1, sB);
    }

    // ---- online softmax (row stats replicated per 16-lane half) ----
    const float L2E = 1.4426950408889634f;
    float alpha[8];
#pragma unroll
    for (int r = 0; r < 8; ++r) {
      float mx = fmaxf(sA[r], sB[r]);
      mx = fmaxf(mx, __shfl_xor(mx, 1, 32));
      mx = fmaxf(mx, __shfl_xor(mx, 2, 32));
      mx = fmaxf(mx, __shfl_xor(mx, 4, 32));
      mx = fmaxf(mx, __shfl_xor(mx, 8, 32));
      const float mn = fmaxf(mrow[r], mx);
      const float al = exp2f((mrow[r] - mn) * L2E);
      const float p0 = exp2f((sA[r] - mn) * L2E);
      const float p1 = exp2f((sB[r] - mn) * L2E);
      float rs = p0 + p1;
      rs += __shfl_xor(rs, 1, 32);
      rs += __shfl_xor(rs, 2, 32);
      rs += __shfl_xor(rs, 4, 32);
      rs += __shfl_xor(rs, 8, 32);
      lrow[r] = lrow[r] * al + rs;
      mrow[r] = mn;
      alpha[r] = al;
      sA[r] = p0;
      sB[r] = p1;
    }

    // ---- re-layout P (C-format) -> A-fragment format through per-wave LDS
    //      (DS ops are in-order within a wave; no barrier needed) ----
#pragma unroll
    for (int r = 0; r < 8; ++r) {
      const int row = r + (hh << 3);
      pbuf[w][row][lane15]      = (_Float16)sA[r];
      pbuf[w][row][lane15 + 16] = (_Float16)sB[r];
    }
    v16h pf;
#pragma unroll
    for (int j = 0; j < 8; ++j) {
      const int k0 = ak(j, hh);
      const h2 pp = *(const h2*)&pbuf[w][lane15][k0];
      pf[2 * j] = pp.x; pf[2 * j + 1] = pp.y;
    }

    // ---- O = O*alpha + P * V (16 column tiles of 16) ----
#pragma unroll
    for (int t = 0; t < 16; ++t) {
#pragma unroll
      for (int r = 0; r < 8; ++r) ot[t][r] *= alpha[r];
      v16h vf;
#pragma unroll
      for (int j = 0; j < 8; ++j) {
        const int k0 = ak(j, hh);
        vf[2 * j]     = kvs[buf][1][k0][(t << 4) + lane15];
        vf[2 * j + 1] = kvs[buf][1][k0 + 1][(t << 4) + lane15];
      }
      ot[t] = wmma_f16(pf, vf, ot[t]);
    }
    __syncthreads();   // everyone done reading buf before it is re-staged
  }

  float inv[8];
#pragma unroll
  for (int r = 0; r < 8; ++r) inv[r] = 1.0f / lrow[r];
#pragma unroll
  for (int t = 0; t < 16; ++t) {
#pragma unroll
    for (int r = 0; r < 8; ++r) {
      const int row = rbase + r + (hh << 3);
      O[row * NO + hoff + (t << 4) + lane15] = (_Float16)(ot[t][r] * inv[r]);
    }
  }
}

// ---------------------------------------------------------------------------
// Kernel 3: unified = O (8192x2048 f16) * Wu^T + bu, fused with x2 residual.
// Each wave computes a 16x64 strip (A fragment reused 4x); grid = (512, 4).
// ---------------------------------------------------------------------------
__global__ __launch_bounds__(32) void unify(const _Float16* __restrict__ O,
                                            const float* __restrict__ Wu,
                                            const float* __restrict__ bu,
                                            const float* __restrict__ x,
                                            float* __restrict__ y) {
  const int l = threadIdx.x;
  const int m0 = blockIdx.x << 4;
  const int c0 = blockIdx.y << 6;
  const int hh = l >> 4;
  const int lane15 = l & 15;
  v8f acc[4];
  const v8f vzero = {};
#pragma unroll
  for (int t = 0; t < 4; ++t) acc[t] = vzero;

  for (int o0 = 0; o0 < NO; o0 += 32) {
    v16h a;
#pragma unroll
    for (int j = 0; j < 8; ++j) {
      const int k0 = o0 + ak(j, hh);
      const h2 aa = *(const h2*)&O[(m0 + lane15) * NO + k0];
      a[2 * j] = aa.x; a[2 * j + 1] = aa.y;
    }
#pragma unroll
    for (int t = 0; t < 4; ++t) {
      v16h b;
#pragma unroll
      for (int j = 0; j < 8; ++j) {
        const int k0 = o0 + ak(j, hh);
        const float* bp = &Wu[(c0 + (t << 4) + lane15) * NO + k0];
        b[2 * j]     = (_Float16)bp[0];
        b[2 * j + 1] = (_Float16)bp[1];
      }
      acc[t] = wmma_f16(a, b, acc[t]);
    }
  }
#pragma unroll
  for (int t = 0; t < 4; ++t) {
    const int col = c0 + (t << 4) + lane15;
    const float bias = bu[col];
#pragma unroll
    for (int r = 0; r < 8; ++r) {
      const int row = m0 + r + (hh << 3);
      y[row * CC + EE + col] = acc[t][r] + bias + x[row * CC + EE + col];
    }
  }
}

// ---------------------------------------------------------------------------
// Kernel 4: passthrough half of the coupling block: y[..., :256] = x1.
// ---------------------------------------------------------------------------
__global__ __launch_bounds__(256) void copy_x1(const float* __restrict__ x,
                                               float* __restrict__ y) {
  const int idx = blockIdx.x * 256 + threadIdx.x;  // float4 index, 8192*64
  const int row = idx >> 6;
  const int c = (idx & 63) << 2;
  *(float4*)&y[row * CC + c] = *(const float4*)&x[row * CC + c];
}

extern "C" void kernel_launch(void* const* d_in, const int* in_sizes, int n_in,
                              void* d_out, int out_size, void* d_ws,
                              size_t ws_size, hipStream_t stream) {
  const float* x  = (const float*)d_in[0];
  const float* Wq = (const float*)d_in[1];
  const float* Wk = (const float*)d_in[2];
  const float* Wv = (const float*)d_in[3];
  const float* Wu = (const float*)d_in[4];
  const float* bu = (const float*)d_in[5];
  float* y = (float*)d_out;

  // workspace: Q, K, V, O each 8192*2048 f16 = 32 MB (128 MB total)
  const size_t NQK = (size_t)MROWS * NO;
  _Float16* ws = (_Float16*)d_ws;
  _Float16* Q = ws;
  _Float16* K = ws + NQK;
  _Float16* V = ws + 2 * NQK;
  _Float16* O = ws + 3 * NQK;

  hipLaunchKernelGGL(qkv_proj, dim3(MROWS / 16, 3 * (NO / 64)), dim3(32), 0,
                     stream, x, Wq, Wk, Wv, Q, K, V);
  hipLaunchKernelGGL(attn, dim3(BB * HH * (TT / 64)), dim3(128), 0, stream, Q,
                     K, V, O);
  hipLaunchKernelGGL(unify, dim3(MROWS / 16, EE / 64), dim3(32), 0, stream, O,
                     Wu, bu, x, y);
  hipLaunchKernelGGL(copy_x1, dim3(MROWS * 64 / 256), dim3(256), 0, stream, x,
                     y);
}
